// lf_85925115723967
// MI455X (gfx1250) — matrix operand order repaired and checked by reading, not compiled, not run on hardware
//
#include <hip/hip_runtime.h>
#include <hip/hip_bf16.h>

// ---------------------------------------------------------------------------
// Types / helpers
// ---------------------------------------------------------------------------
typedef __attribute__((ext_vector_type(16))) __bf16 bf16x16;
typedef __attribute__((ext_vector_type(8)))  float  f32x8;
typedef __attribute__((ext_vector_type(4)))  unsigned int u32x4;
typedef __attribute__((ext_vector_type(8)))  int    i32x8;
typedef __attribute__((ext_vector_type(4)))  int    i32x4;

__device__ __forceinline__ unsigned short f2bf(float f) {
  unsigned int u = __float_as_uint(f);
  u += 0x7FFFu + ((u >> 16) & 1u);          // round-to-nearest-even
  return (unsigned short)(u >> 16);
}

union BFrag { uint4 u[2]; bf16x16 v; };

// A-operand style: two separate 8-element (16B) chunks
__device__ __forceinline__ bf16x16 frag_2x8(const unsigned short* p0,
                                            const unsigned short* p1) {
  BFrag r;
  r.u[0] = *(const uint4*)p0;
  r.u[1] = *(const uint4*)p1;
  return r.v;
}
// B-operand style: one contiguous 16-element (32B) chunk
__device__ __forceinline__ bf16x16 frag_16(const unsigned short* p) {
  BFrag r;
  r.u[0] = *(const uint4*)p;
  r.u[1] = *(const uint4*)(p + 8);
  return r.v;
}

__device__ __forceinline__ f32x8 wmma_bf16(bf16x16 a, bf16x16 b, f32x8 c) {
  return __builtin_amdgcn_wmma_f32_16x16x32_bf16(false, a, false, b,
                                                 (short)0, c, false, false);
}

__device__ __forceinline__ f32x8 zero8() {
  f32x8 z;
#pragma unroll
  for (int i = 0; i < 8; ++i) z[i] = 0.0f;
  return z;
}

// ---------------------------------------------------------------------------
// Tensor Data Mover: 2-D tile load (bf16 elements) global -> LDS.
// Builds the D# descriptor per CDNA5 ISA:
//   group0: [1:0]=count=1, [63:32]=lds_addr, [120:64]=global_addr, [127:126]=2
//   group1: data_size=1 (2B), tensor_dim0/1, tile_dim0/1, tensor_dim0_stride
// Issue from ONE wave only (EXEC is ignored; instruction is per-wave).
// Completion tracked with TENSORcnt -> s_wait_tensorcnt.
// ---------------------------------------------------------------------------
__device__ __forceinline__ void tdm_load_2d(unsigned lds_off, const void* gptr,
                                            unsigned tile_w, unsigned tile_h,
                                            unsigned row_stride_elems) {
  unsigned long long ga = (unsigned long long)(uintptr_t)gptr;
  u32x4 g0;
  g0[0] = 1u;                                               // count=1, user D#
  g0[1] = lds_off;                                          // lds_addr (bytes)
  g0[2] = (unsigned)ga;                                     // global_addr lo
  g0[3] = (unsigned)((ga >> 32) & 0x01FFFFFFu) | (2u << 30);// ga hi | type=2
  i32x8 g1;
  g1[0] = 0x10000;                    // workgroup_mask=0, data_size=1 (2 bytes)
  g1[1] = (int)(tile_w << 16);        // tensor_dim0[15:0] @ bits 63:48
  g1[2] = (int)(tile_h << 16);        // tensor_dim1[15:0] @ bits 95:80
  g1[3] = (int)(tile_w << 16);        // tile_dim0 @ bits 127:112
  g1[4] = (int)tile_h;                // tile_dim1 @ bits 143:128 (tile_dim2=0)
  g1[5] = (int)row_stride_elems;      // tensor_dim0_stride[31:0]
  g1[6] = 0;                          // stride hi, tensor_dim1_stride lo
  g1[7] = 0;
  i32x4 gz;
  gz[0] = 0; gz[1] = 0; gz[2] = 0; gz[3] = 0;
#if defined(__clang_major__) && (__clang_major__ >= 23)
  i32x8 gz8;
#pragma unroll
  for (int i = 0; i < 8; ++i) gz8[i] = 0;
  __builtin_amdgcn_tensor_load_to_lds(g0, g1, gz, gz, gz8, 0);
#else
  __builtin_amdgcn_tensor_load_to_lds(g0, g1, gz, gz, 0);
#endif
}

// LDS raw offset: flat LDS addresses map addr[31:0] directly to LDS bytes.
__device__ __forceinline__ unsigned lds_off_of(const void* p) {
  return (unsigned)(uintptr_t)p;
}

// ---------------------------------------------------------------------------
// fp32 -> bf16 conversion kernels
// ---------------------------------------------------------------------------
__global__ void lf_cvt_bf16(const float* __restrict__ src,
                            unsigned short* __restrict__ dst, int n) {
  int i = blockIdx.x * blockDim.x + threadIdx.x;
  if (i < n) dst[i] = f2bf(src[i]);
}

// P is [128][32] fp32 -> PT [32][128] bf16
__global__ void lf_transpose_p(const float* __restrict__ P,
                               unsigned short* __restrict__ PT) {
  int i = blockIdx.x * blockDim.x + threadIdx.x;   // 4096 threads
  int d = i >> 5, j = i & 31;
  PT[j * 128 + d] = f2bf(P[d * 32 + j]);
}

// ---------------------------------------------------------------------------
// Tiled WMMA GEMM:  C[M,1024] = A[M,1024] @ W[1024,1024]^T + bias
// A,W bf16 row-major (W is [n_out, k] so B-operand reads W rows contiguously).
// Block tile 128x128, BK=64, 256 threads (8 waves), wave tile 32x64.
// Tiles staged by TDM (wave 0), double-buffered so DMA overlaps WMMA.
// ---------------------------------------------------------------------------
template <bool F32OUT>
__global__ __launch_bounds__(256) void lf_gemm_bias(
    const unsigned short* __restrict__ A,
    const unsigned short* __restrict__ W,
    const float* __restrict__ bias,
    void* __restrict__ out) {
  constexpr int N = 1024, K = 1024, BK = 64;
  __shared__ unsigned short As[2][128][BK];
  __shared__ unsigned short Ws[2][128][BK];

  const int m0   = blockIdx.y * 128;
  const int n0   = blockIdx.x * 128;
  const int lane = threadIdx.x & 31;
  const int w    = threadIdx.x >> 5;
  const int rBase = (w & 3) * 32;   // 4 wave-rows of 32
  const int cBase = (w >> 2) * 64;  // 2 wave-cols of 64

  const unsigned ldsA[2] = {lds_off_of(&As[0][0][0]), lds_off_of(&As[1][0][0])};
  const unsigned ldsW[2] = {lds_off_of(&Ws[0][0][0]), lds_off_of(&Ws[1][0][0])};

  __builtin_prefetch(&bias[n0 + (threadIdx.x & 127)], 0, 1);

  f32x8 acc[2][4];
#pragma unroll
  for (int i = 0; i < 2; ++i)
#pragma unroll
    for (int j = 0; j < 4; ++j) acc[i][j] = zero8();

  if (w == 0) {  // prime first tile pair via Tensor Data Mover
    tdm_load_2d(ldsA[0], A + (size_t)m0 * K, BK, 128, K);
    tdm_load_2d(ldsW[0], W + (size_t)n0 * K, BK, 128, K);
  }

  int buf = 0;
  for (int kb = 0; kb < K; kb += BK) {
    if (w == 0) __builtin_amdgcn_s_wait_tensorcnt(0);
    __syncthreads();
    if (w == 0 && (kb + BK) < K) {  // prefetch next tiles into other buffer
      tdm_load_2d(ldsA[buf ^ 1], A + (size_t)m0 * K + kb + BK, BK, 128, K);
      tdm_load_2d(ldsW[buf ^ 1], W + (size_t)n0 * K + kb + BK, BK, 128, K);
    }

    const unsigned short(*Asb)[BK] = As[buf];
    const unsigned short(*Wsb)[BK] = Ws[buf];
#pragma unroll
    for (int ks = 0; ks < BK; ks += 32) {
      bf16x16 af[2];
#pragma unroll
      for (int i = 0; i < 2; ++i) {
        int row = rBase + i * 16 + (lane & 15);
        int k0  = ks + ((lane >> 4) << 4);
        af[i] = frag_2x8(&Asb[row][k0], &Asb[row][k0 + 8]);
      }
#pragma unroll
      for (int j = 0; j < 4; ++j) {
        int col = cBase + j * 16 + (lane & 15);
        int k0  = ks + ((lane >> 4) << 4);
        bf16x16 bf_ = frag_16(&Wsb[col][k0]);
#pragma unroll
        for (int i = 0; i < 2; ++i) acc[i][j] = wmma_bf16(af[i], bf_, acc[i][j]);
      }
    }
    __syncthreads();
    buf ^= 1;
  }

#pragma unroll
  for (int i = 0; i < 2; ++i)
#pragma unroll
    for (int j = 0; j < 4; ++j) {
      int col  = n0 + cBase + j * 16 + (lane & 15);
      float bv = bias[col];
#pragma unroll
      for (int v = 0; v < 8; ++v) {
        int row   = m0 + rBase + i * 16 + v + ((lane >> 4) << 3);
        float val = acc[i][j][v] + bv;
        if (F32OUT)
          ((float*)out)[(size_t)row * N + col] = val;
        else
          ((unsigned short*)out)[(size_t)row * N + col] = f2bf(val);
      }
    }
}

// ---------------------------------------------------------------------------
// Fused Linformer attention per (b,h). 512 workgroups, 256 threads, 128KB LDS.
//   K_proj[j,t]  = sum_d Kh[t,d] * P[d,j]    (stored j-major for scores B-op)
//   V_projT[t,j] = sum_d Vh[t,d] * P[d,j]    (stored t-major for out B-op)
//   scores[s,j]  = sum_t Qh[s,t] * K_proj[j,t] / sqrt(128);  softmax over j
//   ctx[s,t']    = sum_j attn[s,j] * V_projT[t',j]
// Q/K/V/PT tiles staged via TDM.
// ---------------------------------------------------------------------------
__global__ __launch_bounds__(256) void lf_linf_attn(
    const unsigned short* __restrict__ Qg,
    const unsigned short* __restrict__ Kg,
    const unsigned short* __restrict__ Vg,
    const unsigned short* __restrict__ PTg,  // [32][128] bf16
    unsigned short* __restrict__ Ctx) {
  extern __shared__ char smem[];
  auto Qs = (unsigned short(*)[128])(smem);             // 32 KB
  auto Ks = (unsigned short(*)[128])(smem + 32768);     // 32 KB
  auto Vs = (unsigned short(*)[128])(smem + 65536);     // 32 KB
  auto Ps = (unsigned short(*)[128])(smem + 98304);     //  8 KB  PT[32][128]
  auto Kp = (unsigned short(*)[128])(smem + 106496);    //  8 KB  K_proj[32][128]
  auto Vp = (unsigned short(*)[32])(smem + 114688);     //  8 KB  V_projT[128][32]
  auto At = (unsigned short(*)[32])(smem + 122880);     //  8 KB  attn[128][32]

  const int b    = blockIdx.x >> 3;
  const int h    = blockIdx.x & 7;
  const int lane = threadIdx.x & 31;
  const int w    = threadIdx.x >> 5;
  const size_t base = (size_t)b * 128 * 1024 + (size_t)h * 128;

  // ---- stage Q/K/V head tiles + PT into LDS via Tensor Data Mover ----
  if (w == 0) {
    const unsigned l0 = lds_off_of(smem);
    tdm_load_2d(l0 +     0u, Qg + base, 128, 128, 1024);
    tdm_load_2d(l0 + 32768u, Kg + base, 128, 128, 1024);
    tdm_load_2d(l0 + 65536u, Vg + base, 128, 128, 1024);
    tdm_load_2d(l0 + 98304u, PTg,       128,  32,  128);
    __builtin_amdgcn_s_wait_tensorcnt(0);
  }
  __syncthreads();

  const int m0 = w * 16;  // each wave owns 16 rows of the 128-row tile

  // ---- projections: Kh x P and Vh x P (M=128, N=32, K=128) ----
  {
    f32x8 ak[2] = {zero8(), zero8()};
    f32x8 av[2] = {zero8(), zero8()};
#pragma unroll
    for (int kb = 0; kb < 128; kb += 32) {
      int rowA = m0 + (lane & 15);
      int k0a  = kb + ((lane >> 4) << 4);
      bf16x16 aK = frag_2x8(&Ks[rowA][k0a], &Ks[rowA][k0a + 8]);
      bf16x16 aV = frag_2x8(&Vs[rowA][k0a], &Vs[rowA][k0a + 8]);
      int k0b = kb + ((lane >> 4) << 4);
#pragma unroll
      for (int nf = 0; nf < 2; ++nf) {
        bf16x16 bP = frag_16(&Ps[nf * 16 + (lane & 15)][k0b]);
        ak[nf] = wmma_bf16(aK, bP, ak[nf]);
        av[nf] = wmma_bf16(aV, bP, av[nf]);
      }
    }
#pragma unroll
    for (int nf = 0; nf < 2; ++nf) {
      int ncol = nf * 16 + (lane & 15);
#pragma unroll
      for (int v = 0; v < 8; ++v) {
        int trow = m0 + v + ((lane >> 4) << 3);
        Kp[ncol][trow] = f2bf(ak[nf][v]);  // transposed store -> [j][t]
        Vp[trow][ncol] = f2bf(av[nf][v]);  // direct store     -> [t][j]
      }
    }
  }
  __syncthreads();

  // ---- scores (M=128, N=32, K=128) + in-register softmax over 32 ----
  {
    f32x8 sc[2] = {zero8(), zero8()};
#pragma unroll
    for (int kb = 0; kb < 128; kb += 32) {
      int rowA = m0 + (lane & 15);
      int k0a  = kb + ((lane >> 4) << 4);
      bf16x16 aQ = frag_2x8(&Qs[rowA][k0a], &Qs[rowA][k0a + 8]);
      int k0b = kb + ((lane >> 4) << 4);
#pragma unroll
      for (int nf = 0; nf < 2; ++nf) {
        bf16x16 bK = frag_16(&Kp[nf * 16 + (lane & 15)][k0b]);
        sc[nf] = wmma_bf16(aQ, bK, sc[nf]);
      }
    }
    const float scale = 0.08838834764831845f;  // 1/sqrt(128)
#pragma unroll
    for (int v = 0; v < 8; ++v) {
      float s0 = sc[0][v] * scale;
      float s1 = sc[1][v] * scale;
      float mx = fmaxf(s0, s1);
#pragma unroll
      for (int d = 1; d < 16; d <<= 1) mx = fmaxf(mx, __shfl_xor(mx, d, 32));
      float e0 = __expf(s0 - mx), e1 = __expf(s1 - mx);
      float sm = e0 + e1;
#pragma unroll
      for (int d = 1; d < 16; d <<= 1) sm += __shfl_xor(sm, d, 32);
      float inv = 1.0f / sm;
      int row = m0 + v + ((lane >> 4) << 3);
      At[row][lane & 15]        = f2bf(e0 * inv);
      At[row][16 + (lane & 15)] = f2bf(e1 * inv);
    }
  }
  __syncthreads();

  // ---- out = attn(128x32) @ V_projT-as-B (K=32, N=128) ----
  {
    int rowA = m0 + (lane & 15);
    int k0a  = (lane >> 4) << 4;
    bf16x16 aA = frag_2x8(&At[rowA][k0a], &At[rowA][8 + k0a]);
    int k0b = (lane >> 4) << 4;
#pragma unroll
    for (int nt = 0; nt < 8; ++nt) {
      bf16x16 bV = frag_16(&Vp[nt * 16 + (lane & 15)][k0b]);
      f32x8 o = wmma_bf16(aA, bV, zero8());
      int col = nt * 16 + (lane & 15);
#pragma unroll
      for (int v = 0; v < 8; ++v) {
        int row = m0 + v + ((lane >> 4) << 3);
        Ctx[base + (size_t)row * 1024 + col] = f2bf(o[v]);
      }
    }
  }
}

// ---------------------------------------------------------------------------
// Launch
// ---------------------------------------------------------------------------
extern "C" void kernel_launch(void* const* d_in, const int* in_sizes, int n_in,
                              void* d_out, int out_size, void* d_ws,
                              size_t ws_size, hipStream_t stream) {
  (void)in_sizes; (void)n_in; (void)out_size; (void)ws_size;
  const float* x  = (const float*)d_in[0];
  const float* Wq = (const float*)d_in[1];
  const float* bq = (const float*)d_in[2];
  const float* Wk = (const float*)d_in[3];
  const float* bk = (const float*)d_in[4];
  const float* Wv = (const float*)d_in[5];
  const float* bv = (const float*)d_in[6];
  const float* P  = (const float*)d_in[7];
  const float* Wo = (const float*)d_in[8];
  const float* bo = (const float*)d_in[9];
  float* out = (float*)d_out;

  const size_t M = 8192, E = 1024;
  char* ws = (char*)d_ws;
  size_t off = 0;
  auto carve = [&](size_t bytes) -> char* {
    char* p = ws + off;
    off += (bytes + 255) & ~(size_t)255;
    return p;
  };
  unsigned short* xb  = (unsigned short*)carve(M * E * 2);
  unsigned short* Wqb = (unsigned short*)carve(E * E * 2);
  unsigned short* Wkb = (unsigned short*)carve(E * E * 2);
  unsigned short* Wvb = (unsigned short*)carve(E * E * 2);
  unsigned short* Wob = (unsigned short*)carve(E * E * 2);
  unsigned short* PTb = (unsigned short*)carve(32 * 128 * 2);
  unsigned short* Qb  = (unsigned short*)carve(M * E * 2);
  unsigned short* Kb  = (unsigned short*)carve(M * E * 2);
  unsigned short* Vb  = (unsigned short*)carve(M * E * 2);
  unsigned short* Cb  = (unsigned short*)carve(M * E * 2);

  lf_cvt_bf16<<<(int)(M * E / 256), 256, 0, stream>>>(x, xb, (int)(M * E));
  lf_cvt_bf16<<<(int)(E * E / 256), 256, 0, stream>>>(Wq, Wqb, (int)(E * E));
  lf_cvt_bf16<<<(int)(E * E / 256), 256, 0, stream>>>(Wk, Wkb, (int)(E * E));
  lf_cvt_bf16<<<(int)(E * E / 256), 256, 0, stream>>>(Wv, Wvb, (int)(E * E));
  lf_cvt_bf16<<<(int)(E * E / 256), 256, 0, stream>>>(Wo, Wob, (int)(E * E));
  lf_transpose_p<<<16, 256, 0, stream>>>(P, PTb);

  dim3 gg(8, 64);  // N/128, M/128
  lf_gemm_bias<false><<<gg, 256, 0, stream>>>(xb, Wqb, bq, (void*)Qb);
  lf_gemm_bias<false><<<gg, 256, 0, stream>>>(xb, Wkb, bk, (void*)Kb);
  lf_gemm_bias<false><<<gg, 256, 0, stream>>>(xb, Wvb, bv, (void*)Vb);

  lf_linf_attn<<<512, 256, 131072, stream>>>(Qb, Kb, Vb, PTb, Cb);

  lf_gemm_bias<true><<<gg, 256, 0, stream>>>(Cb, Wob, bo, (void*)out);
}
